// F2TDReid_90872918049494
// MI455X (gfx1250) — matrix operand order repaired and checked by reading, not compiled, not run on hardware
//
#include <hip/hip_runtime.h>
#include <hip/hip_bf16.h>

// ---------------------------------------------------------------------------
// MI455X (gfx1250) implementation.
//   - Big GEMMs (QK^T with K=8 padded to 32, and P*V with K=1024) run on
//     v_wmma_f32_16x16x32_f16 (wave32 WMMA, f32 accumulate).
//   - Softmax is done without max-subtraction (scores ~ +-3, exp is safe),
//     single pass: row sums come from one extra WMMA vs an all-ones B frag.
//   - Workspace layout (needs ~34 MB of d_ws):
//       qT   : [64][1024][32] f16 (K-padded, zeros for d>=8)   4 MB
//       kT   : [64][1024][32] f16                              4 MB
//       vh   : [64][64][1024] f16                              8 MB
//       res  : [64][64][1024] f32                             16 MB
//       emb  : [64][128] f32, lam_ws: 64 f32, loss: 1 f32
// ---------------------------------------------------------------------------

typedef __attribute__((ext_vector_type(16))) _Float16 v16h;
typedef __attribute__((ext_vector_type(8)))  _Float16 v8h;
typedef __attribute__((ext_vector_type(8)))  float    v8f;

#define WS_QT   0
#define WS_KT   (4u * 1024u * 1024u)
#define WS_VH   (8u * 1024u * 1024u)
#define WS_RES  (16u * 1024u * 1024u)
#define WS_EMB  (32u * 1024u * 1024u)
#define WS_LAM  (WS_EMB + 64u * 128u * 4u)
#define WS_LOSS (WS_LAM + 256u)

__device__ inline v16h ld16h(const _Float16* p0, const _Float16* p1) {
    v8h lo = *(const v8h*)p0;
    v8h hi = *(const v8h*)p1;
    return __builtin_shufflevector(lo, hi, 0, 1, 2, 3, 4, 5, 6, 7,
                                   8, 9, 10, 11, 12, 13, 14, 15);
}

// --------------------------------------------------------------------------
// K0: zero the loss accumulator
// --------------------------------------------------------------------------
__global__ void init_kernel(float* __restrict__ loss) {
    if (threadIdx.x == 0) loss[0] = 0.0f;
}

// --------------------------------------------------------------------------
// K1: per-batch QKV projection (K=64 dots, VALU). Writes WMMA-friendly
//     layouts: qT/kT K-contiguous padded to 32, v row-major [c][n].
// --------------------------------------------------------------------------
__global__ __launch_bounds__(256)
void qkv_kernel(const float* __restrict__ x,
                const float* __restrict__ wq_s, const float* __restrict__ bq_s,
                const float* __restrict__ wk_s, const float* __restrict__ bk_s,
                const float* __restrict__ wv_s, const float* __restrict__ bv_s,
                const float* __restrict__ wq_t, const float* __restrict__ bq_t,
                const float* __restrict__ wk_t, const float* __restrict__ bk_t,
                const float* __restrict__ wv_t, const float* __restrict__ bv_t,
                _Float16* __restrict__ qT, _Float16* __restrict__ kT,
                _Float16* __restrict__ vh) {
    __shared__ float wq_l[512], wk_l[512], wv_l[4096];
    __shared__ float bq_l[8], bk_l[8], bv_l[64];
    const int b = blockIdx.x;
    const bool src = (b < 32);
    const float* wq = src ? wq_s : wq_t;
    const float* wk = src ? wk_s : wk_t;
    const float* wv = src ? wv_s : wv_t;
    const float* bq = src ? bq_s : bq_t;
    const float* bk = src ? bk_s : bk_t;
    const float* bv = src ? bv_s : bv_t;

    for (int i = threadIdx.x; i < 512; i += 256) { wq_l[i] = wq[i]; wk_l[i] = wk[i]; }
    for (int i = threadIdx.x; i < 4096; i += 256) wv_l[i] = wv[i];
    if (threadIdx.x < 8)  { bq_l[threadIdx.x] = bq[threadIdx.x]; bk_l[threadIdx.x] = bk[threadIdx.x]; }
    if (threadIdx.x < 64) bv_l[threadIdx.x] = bv[threadIdx.x];
    __syncthreads();

    const float* xb = x + ((size_t)b << 16);   // x[b][c][n], c stride 1024
    for (int col = 0; col < 4; ++col) {
        const int n = threadIdx.x + col * 256;
        float xc[64];
#pragma unroll
        for (int c = 0; c < 64; ++c) xc[c] = xb[(c << 10) + n];

        _Float16* qrow = qT + ((size_t)((b << 10) + n)) * 32;
        _Float16* krow = kT + ((size_t)((b << 10) + n)) * 32;
#pragma unroll
        for (int d = 0; d < 8; ++d) {
            float aq = bq_l[d], ak2 = bk_l[d];
#pragma unroll
            for (int c = 0; c < 64; ++c) {
                aq  += wq_l[(d << 6) + c] * xc[c];
                ak2 += wk_l[(d << 6) + c] * xc[c];
            }
            qrow[d] = (_Float16)aq;
            krow[d] = (_Float16)ak2;
        }
#pragma unroll
        for (int d = 8; d < 32; ++d) { qrow[d] = (_Float16)0.0f; krow[d] = (_Float16)0.0f; }

        for (int co = 0; co < 64; ++co) {
            float acc = bv_l[co];
#pragma unroll
            for (int c = 0; c < 64; ++c) acc += wv_l[(co << 6) + c] * xc[c];
            vh[(((size_t)(b * 64 + co)) << 10) + n] = (_Float16)acc;
        }
    }
}

// --------------------------------------------------------------------------
// K2: fused attention. One wave = one 16-row n-tile of one batch.
//     Per 32-wide m-chunk: 2 WMMA for scores, exp via LDS bounce, then
//     4 WMMA for P*V (c-tiles) + 1 WMMA vs all-ones B for row sums.
// --------------------------------------------------------------------------
__global__ __launch_bounds__(256)
void attn_kernel(const float* __restrict__ x,
                 const _Float16* __restrict__ qT,
                 const _Float16* __restrict__ kT,
                 const _Float16* __restrict__ vh,
                 float* __restrict__ res,
                 const float* __restrict__ a_s, const float* __restrict__ a_t,
                 float* __restrict__ loss) {
    __shared__ _Float16 p_smem[8 * 512];   // 8 waves x [16 rows][32 cols] f16
    const int tid  = threadIdx.x;
    const int lane = tid & 31;
    const int wv   = tid >> 5;
    const int l16  = lane & 15;
    const int half = lane >> 4;
    const int b    = blockIdx.x >> 3;
    const int nbase = ((blockIdx.x & 7) * 8 + wv) * 16;

    const _Float16* qb = qT + (((size_t)b) << 10) * 32;
    const _Float16* kb = kT + (((size_t)b) << 10) * 32;
    const _Float16* vb = vh + (((size_t)b) << 6) * 1024;

    const int ak = half ? 16 : 0;   // A-frag K offset per lane half
    const int bk = half ? 16 : 0;   // B-frag K offset per lane half

    const _Float16* qp = qb + (size_t)(nbase + l16) * 32 + ak;
    v16h a_q = ld16h(qp, qp + 8);

    v16h ones;
#pragma unroll
    for (int i = 0; i < 16; ++i) ones[i] = (_Float16)1.0f;

    v8f accv[4] = {};
    v8f accS = {};
    _Float16* pw = p_smem + wv * 512;

    for (int mt = 0; mt < 1024; mt += 32) {
        // scores: A = q rows (16x32, d padded), B = k^T rows (K-contiguous)
        const _Float16* kp0 = kb + (size_t)(mt + l16) * 32 + bk;
        const _Float16* kp1 = kb + (size_t)(mt + 16 + l16) * 32 + bk;
        v16h b_k0 = ld16h(kp0, kp0 + 8);
        v16h b_k1 = ld16h(kp1, kp1 + 8);
        v8f s0 = {}, s1 = {};
        s0 = __builtin_amdgcn_wmma_f32_16x16x32_f16(false, a_q, false, b_k0,
                                                    (short)0, s0, false, false);
        s1 = __builtin_amdgcn_wmma_f32_16x16x32_f16(false, a_q, false, b_k1,
                                                    (short)0, s1, false, false);
        // exp + C-frag -> A-frag relayout through per-wave LDS tile
        const int rb = half * 8;
#pragma unroll
        for (int j = 0; j < 8; ++j) {
            pw[(rb + j) * 32 + l16]      = (_Float16)__expf(s0[j]);
            pw[(rb + j) * 32 + 16 + l16] = (_Float16)__expf(s1[j]);
        }
        const _Float16* pr = pw + l16 * 32 + ak;
        v16h a_p = ld16h(pr, pr + 8);

        // P*V over 4 c-tiles; B = v rows (already K-contiguous in m)
#pragma unroll
        for (int t = 0; t < 4; ++t) {
            const _Float16* vp = vb + (((size_t)(t * 16 + l16)) << 10) + mt + bk;
            v16h b_v = ld16h(vp, vp + 8);
            accv[t] = __builtin_amdgcn_wmma_f32_16x16x32_f16(false, a_p, false, b_v,
                                                             (short)0, accv[t], false, false);
        }
        // row sums of exp(S): every column of accS equals the row sum
        accS = __builtin_amdgcn_wmma_f32_16x16x32_f16(false, a_p, false, ones,
                                                      (short)0, accS, false, false);
    }

    const float aval = (b < 32) ? a_s[0] : a_t[0];
    float inv[8];
#pragma unroll
    for (int j = 0; j < 8; ++j) inv[j] = 1.0f / accS[j];

    float lsum = 0.0f;
    const int n0 = nbase + half * 8;   // C-frag rows j+8*half are consecutive
#pragma unroll
    for (int t = 0; t < 4; ++t) {
        const int c = t * 16 + l16;
        const size_t base = ((((size_t)b) * 64 + c) << 10) + n0;
        const float* xr = x + base;
        float* rr = res + base;
#pragma unroll
        for (int j = 0; j < 8; ++j) {
            float av = accv[t][j] * inv[j];
            float xv = xr[j];
            float r  = aval * av + xv;
            rr[j] = r;
            float d = aval * r - xv;
            lsum += d * d;
        }
    }
#pragma unroll
    for (int off = 16; off; off >>= 1) lsum += __shfl_xor(lsum, off);
    if (lane == 0) atomicAdd(loss, lsum);
}

// --------------------------------------------------------------------------
// K3: per-image channel avg/max over 1024 spatial -> emb[img][0:64]=avg,
//     emb[img][64:128]=max. 4 threads per channel, combined via shuffles.
// --------------------------------------------------------------------------
__global__ __launch_bounds__(256)
void embed_kernel(const float* __restrict__ res, float* __restrict__ emb) {
    const int img = blockIdx.x;            // 0..63
    const int t = threadIdx.x;
    const int ch = t >> 2, q = t & 3;
    const float* p = res + (((size_t)(img * 64 + ch)) << 10) + q * 256;
    float s = 0.0f, m = -3.4e38f;
    for (int i = 0; i < 256; ++i) { float v = p[i]; s += v; m = fmaxf(m, v); }
    s += __shfl_xor(s, 1); s += __shfl_xor(s, 2);
    m = fmaxf(m, __shfl_xor(m, 1)); m = fmaxf(m, __shfl_xor(m, 2));
    if (q == 0) {
        emb[img * 128 + ch]      = s * (1.0f / 1024.0f);
        emb[img * 128 + 64 + ch] = m;
    }
}

// --------------------------------------------------------------------------
// K4: MLP head. e = fc1(emb_s)+fc1(emb_t); lam = softmax(fc3(fc2(e))).
//     Also writes the two trailing scalars of d_out.
// --------------------------------------------------------------------------
__global__ __launch_bounds__(32)
void head_kernel(const float* __restrict__ emb,
                 const float* __restrict__ w1, const float* __restrict__ b1,
                 const float* __restrict__ w2, const float* __restrict__ b2,
                 const float* __restrict__ w3, const float* __restrict__ b3,
                 const float* __restrict__ loss_acc,
                 float* __restrict__ lam_ws, float* __restrict__ out_tail) {
    __shared__ float e_s[32][64];
    const int i = threadIdx.x;
    if (i < 32) {
        const float* es = emb + i * 128;
        const float* et = emb + (32 + i) * 128;
        for (int o = 0; o < 64; ++o) {
            float acc = 2.0f * b1[o];
            const float* w = w1 + o * 128;
            for (int j = 0; j < 128; ++j) acc += w[j] * (es[j] + et[j]);
            e_s[i][o] = acc;
        }
        float h2[32];
        for (int o = 0; o < 32; ++o) {
            float acc = b2[o];
            const float* w = w2 + o * 64;
            for (int j = 0; j < 64; ++j) acc += w[j] * e_s[i][j];
            h2[o] = acc;
        }
        float h3[2];
        for (int o = 0; o < 2; ++o) {
            float acc = b3[o];
            const float* w = w3 + o * 32;
            for (int j = 0; j < 32; ++j) acc += w[j] * h2[j];
            h3[o] = acc;
        }
        float mx = fmaxf(h3[0], h3[1]);
        float e0 = __expf(h3[0] - mx), e1 = __expf(h3[1] - mx);
        float inv = 1.0f / (e0 + e1);
        float l0 = e0 * inv, l1 = e1 * inv;
        out_tail[i * 2] = l0; out_tail[i * 2 + 1] = l1;
        lam_ws[i * 2] = l0;   lam_ws[i * 2 + 1] = l1;
        if (i == 0) {
            out_tail[64] = 0.0f;                                    // third output
            out_tail[65] = loss_acc[0] * (1.0f / 2097152.0f);       // a_st_loss
        }
    }
}

// --------------------------------------------------------------------------
// K5: out[0:64] = x (x_s and x_t), out[64:96] = lam0*x_s + lam1*x_t.
// --------------------------------------------------------------------------
__global__ __launch_bounds__(256)
void assemble_kernel(const float* __restrict__ x, const float* __restrict__ lam,
                     float* __restrict__ out) {
    const size_t idx = (size_t)blockIdx.x * 256 + threadIdx.x;   // float4 index
    const float4* x4 = (const float4*)x;
    float4* o4 = (float4*)out;
    const size_t NX4 = (size_t)64 * 64 * 1024 / 4;               // 1,048,576
    if (idx < NX4) {
        o4[idx] = x4[idx];
    } else {
        const size_t rel = idx - NX4;
        const int img = (int)(rel >> 14);                        // 16384 f4/img
        const size_t within = rel & 16383;
        const float l0 = lam[img * 2], l1 = lam[img * 2 + 1];
        float4 a = x4[(size_t)img * 16384 + within];
        float4 b = x4[(size_t)(img + 32) * 16384 + within];
        float4 r;
        r.x = l0 * a.x + l1 * b.x; r.y = l0 * a.y + l1 * b.y;
        r.z = l0 * a.z + l1 * b.z; r.w = l0 * a.w + l1 * b.w;
        o4[idx] = r;
    }
}

extern "C" void kernel_launch(void* const* d_in, const int* in_sizes, int n_in,
                              void* d_out, int out_size, void* d_ws, size_t ws_size,
                              hipStream_t stream) {
    const float* x    = (const float*)d_in[0];
    const float* wq_s = (const float*)d_in[1];
    const float* bq_s = (const float*)d_in[2];
    const float* wk_s = (const float*)d_in[3];
    const float* bk_s = (const float*)d_in[4];
    const float* wv_s = (const float*)d_in[5];
    const float* bv_s = (const float*)d_in[6];
    const float* wq_t = (const float*)d_in[7];
    const float* bq_t = (const float*)d_in[8];
    const float* wk_t = (const float*)d_in[9];
    const float* bk_t = (const float*)d_in[10];
    const float* wv_t = (const float*)d_in[11];
    const float* bv_t = (const float*)d_in[12];
    const float* fc1w = (const float*)d_in[13];
    const float* fc1b = (const float*)d_in[14];
    const float* fc2w = (const float*)d_in[15];
    const float* fc2b = (const float*)d_in[16];
    const float* fc3w = (const float*)d_in[17];
    const float* fc3b = (const float*)d_in[18];
    const float* a_s  = (const float*)d_in[19];
    const float* a_t  = (const float*)d_in[20];

    char* ws = (char*)d_ws;
    _Float16* qT  = (_Float16*)(ws + WS_QT);
    _Float16* kT  = (_Float16*)(ws + WS_KT);
    _Float16* vh  = (_Float16*)(ws + WS_VH);
    float* res    = (float*)(ws + WS_RES);
    float* emb    = (float*)(ws + WS_EMB);
    float* lam_ws = (float*)(ws + WS_LAM);
    float* loss   = (float*)(ws + WS_LOSS);

    float* outf = (float*)d_out;
    float* out_tail = outf + (size_t)96 * 64 * 1024;   // 6,291,456

    init_kernel<<<1, 32, 0, stream>>>(loss);
    qkv_kernel<<<64, 256, 0, stream>>>(x, wq_s, bq_s, wk_s, bk_s, wv_s, bv_s,
                                       wq_t, bq_t, wk_t, bk_t, wv_t, bv_t,
                                       qT, kT, vh);
    attn_kernel<<<512, 256, 0, stream>>>(x, qT, kT, vh, res, a_s, a_t, loss);
    embed_kernel<<<64, 256, 0, stream>>>(res, emb);
    head_kernel<<<1, 32, 0, stream>>>(emb, fc1w, fc1b, fc2w, fc2b, fc3w, fc3b,
                                      loss, lam_ws, out_tail);
    assemble_kernel<<<6144, 256, 0, stream>>>(x, lam_ws, outf);
}
